// FixedSparseMultiHead_48593259987330
// MI455X (gfx1250) — matrix operand order repaired and checked by reading, not compiled, not run on hardware
//
#include <hip/hip_runtime.h>

// ---------------- problem constants ----------------
#define BATCH   2
#define SEQ     2048
#define DM      1024
#define NH      16
#define HD      64
#define MTOT    (BATCH * SEQ)      // 4096 tokens
#define WHALF   102                // local window half-width: w=int(2048*0.1)=204 -> 102

typedef __attribute__((ext_vector_type(16))) __bf16 v16bf;
typedef __attribute__((ext_vector_type(8)))  __bf16 v8bf;
typedef __attribute__((ext_vector_type(8)))  float  v8f;
typedef int v4i __attribute__((vector_size(16)));

#if defined(__AMDGCN__) && __has_builtin(__builtin_amdgcn_global_load_async_to_lds_b128)
#define HAVE_ASYNC_LDS 1
#else
#define HAVE_ASYNC_LDS 0
#endif

#if defined(__AMDGCN__) && __has_builtin(__builtin_amdgcn_s_wait_asynccnt)
#define WAIT_ASYNC(n) __builtin_amdgcn_s_wait_asynccnt(n)
#else
#define WAIT_ASYNC(n)
#endif

__device__ __forceinline__ unsigned short f2bf(float f) {
  union { float f; unsigned int u; } v; v.f = f;
  unsigned int u = v.u;
  unsigned int r = u + 0x7FFFu + ((u >> 16) & 1u);   // round-to-nearest-even
  return (unsigned short)(r >> 16);
}

#if HAVE_ASYNC_LDS
typedef __attribute__((address_space(3))) v4i* lds_v4i_t;
typedef __attribute__((address_space(1))) v4i* glb_v4i_t;
__device__ __forceinline__ lds_v4i_t lds_cast(void* p) {
  // generic LDS pointer -> 32-bit LDS offset (aperture: LDS offset is the low 32 address bits)
  return (lds_v4i_t)(unsigned int)(unsigned long long)p;
}
__device__ __forceinline__ glb_v4i_t glob_cast(const void* p) {
  return (glb_v4i_t)(unsigned long long)p;
}
#endif

// ---------------- fp32 -> bf16 conversion ----------------
__global__ void cvt_f32_bf16(const float* __restrict__ src,
                             unsigned short* __restrict__ dst, int n) {
  int i = (blockIdx.x * blockDim.x + threadIdx.x) * 4;
  if (i + 3 < n) {
    float4 f = *(const float4*)(src + i);
    dst[i + 0] = f2bf(f.x);
    dst[i + 1] = f2bf(f.y);
    dst[i + 2] = f2bf(f.z);
    dst[i + 3] = f2bf(f.w);
  } else {
    for (; i < n; ++i) dst[i] = f2bf(src[i]);
  }
}

// ---------------- LDS tile staging ----------------
// A tile: 128 rows x 32 bf16 (8192 B) ; B tile: 64 cols x 32 bf16 (4096 B)
// 256 threads: 2 A-chunks + 1 B-chunk of 16 B each per thread (3 async ops).
__device__ __forceinline__ void stage_tiles(char* abuf, char* bbuf,
                                            const unsigned short* __restrict__ A,
                                            const unsigned short* __restrict__ W,
                                            int mbase, int n0, int k0, int tid) {
#pragma unroll
  for (int t = 0; t < 2; ++t) {
    const int c = tid + t * 256;            // 512 chunks
    const int row = c >> 2, piece = c & 3;  // 4 x 16B per 64B row
    const unsigned short* g = A + (size_t)(mbase + row) * DM + k0 + piece * 8;
    char* l = abuf + c * 16;
#if HAVE_ASYNC_LDS
    __builtin_amdgcn_global_load_async_to_lds_b128(glob_cast(g), lds_cast(l), 0, 0);
#else
    *(uint4*)l = *(const uint4*)g;
#endif
  }
  {
    const int c = tid;                      // 256 chunks
    const int col = c >> 2, piece = c & 3;
    const unsigned short* g = W + (size_t)(n0 + col) * DM + k0 + piece * 8;
    char* l = bbuf + c * 16;
#if HAVE_ASYNC_LDS
    __builtin_amdgcn_global_load_async_to_lds_b128(glob_cast(g), lds_cast(l), 0, 0);
#else
    *(uint4*)l = *(const uint4*)g;
#endif
  }
}

// ---------------- WMMA GEMM: C[m,n] = sum_d A[m,d] * W[n,d] + bias[n] ----------------
// M = 4096, N = K = 1024.  Workgroup (8 waves): 128(M) x 64(N) tile; each wave 16x64.
// K stepped by 32 with a double-buffered async-to-LDS pipeline.
// mode 0: Q -> bf16 [B,H,S,HD]
// mode 1: K -> bf16 [B,H,S,HD]
// mode 2: V -> bf16 [B,H,HD,S]   (transposed for attn*V B-fragments)
// mode 3: out -> f32 [M, N]      (final projection)
__global__ __launch_bounds__(256) void gemm16_bf16(
    const unsigned short* __restrict__ A,
    const unsigned short* __restrict__ W,
    const float* __restrict__ bias,
    void* __restrict__ Cout, int mode) {
  __shared__ __align__(128) char smem[2 * 12288];   // 2 x (A 8KB + B 4KB)

  const int tid  = threadIdx.x;
  const int lane = tid & 31;
  const int wave = tid >> 5;
  const int mb   = blockIdx.x >> 4;         // 32 m-superblocks of 128
  const int nt   = blockIdx.x & 15;         // 16 n-tiles of 64
  const int mbase = mb * 128;
  const int m0    = mbase + wave * 16;
  const int n0    = nt * 64;
  const int row  = lane & 15;
  const int half = lane >> 4;

  char* ab0 = smem;            char* bb0 = smem + 8192;
  char* ab1 = smem + 12288;    char* bb1 = smem + 12288 + 8192;

  v8f acc[4] = {v8f{}, v8f{}, v8f{}, v8f{}};

  stage_tiles(ab0, bb0, A, W, mbase, n0, 0, tid);
  int cur = 0;
  for (int k0 = 0; k0 < DM; k0 += 32) {
    if (k0 + 32 < DM) {
      stage_tiles(cur ? ab0 : ab1, cur ? bb0 : bb1, A, W, mbase, n0, k0 + 32, tid);
      WAIT_ASYNC(3);           // previous stage's 3 ops complete (in-order)
    } else {
      WAIT_ASYNC(0);
    }
    __syncthreads();           // all waves' tiles visible

    const char* ab = cur ? ab1 : ab0;
    const char* bb = cur ? bb1 : bb0;

    // A fragment (16x32): lane half selects K groups {0-15} vs {16-31}
    v8bf alo = *(const v8bf*)(ab + (wave * 16 + row) * 64 + half * 32);
    v8bf ahi = *(const v8bf*)(ab + (wave * 16 + row) * 64 + half * 32 + 16);
    v16bf a;
#pragma unroll
    for (int i = 0; i < 8; ++i) { a[i] = alo[i]; a[8 + i] = ahi[i]; }

#pragma unroll
    for (int i = 0; i < 4; ++i) {
      // B fragment (32x16): lane column, 16 contiguous K at half*16
      v16bf b = *(const v16bf*)(bb + (i * 16 + row) * 64 + half * 32);
      acc[i] = __builtin_amdgcn_wmma_f32_16x16x32_bf16(
          false, a, false, b, (short)0, acc[i], false, false);
    }

    __syncthreads();           // buffer free for next stage
    cur ^= 1;
  }

#pragma unroll
  for (int i = 0; i < 4; ++i) {
    const int n  = n0 + i * 16 + row;       // C/D: N = lane & 15
    const float bv = bias[n];
#pragma unroll
    for (int r = 0; r < 8; ++r) {
      const int m = m0 + r + half * 8;      // C/D: M = r + 8*half
      const float v = acc[i][r] + bv;
      if (mode == 3) {
        ((float*)Cout)[(size_t)m * DM + n] = v;
      } else {
        const int b = m >> 11, s = m & (SEQ - 1);
        const int h = n >> 6,  d = n & (HD - 1);
        size_t idx = (mode == 2)
            ? ((size_t)((b * NH + h) * HD + d) * SEQ + s)    // V^T
            : ((size_t)((b * NH + h) * SEQ + s) * HD + d);   // Q / K
        ((unsigned short*)Cout)[idx] = f2bf(v);
      }
    }
  }
}

// ---------------- sparse local attention ----------------
// One wave per 16-query block.  Dynamic LDS: per wave 16x256 f32 scores + 16x256 bf16 probs.
__global__ __launch_bounds__(64) void attn_local(
    const unsigned short* __restrict__ Q,   // [B,H,S,HD]
    const unsigned short* __restrict__ Km,  // [B,H,S,HD]
    const unsigned short* __restrict__ Vt,  // [B,H,HD,S]
    unsigned short* __restrict__ O) {       // [B,S,DM] bf16
  extern __shared__ char smem[];
  const int lane = threadIdx.x & 31;
  const int wave = threadIdx.x >> 5;
  float*          sc = (float*)(smem + wave * 24576);                 // [16][256]
  unsigned short* pr = (unsigned short*)(smem + wave * 24576 + 16384);// [16][256]

  const int w  = blockIdx.x * 2 + wave;     // 4096 query blocks
  const int bh = w >> 7;                    // 128 q-blocks per (b,h)
  const int qb = w & 127;
  const int b  = bh >> 4, h = bh & 15;
  const int q0 = qb * 16;
  const int row = lane & 15, half = lane >> 4;

  int jlo = q0 - WHALF;        if (jlo < 0) jlo = 0;
  int jhi = q0 + 15 + WHALF;   if (jhi > SEQ - 1) jhi = SEQ - 1;
  const int kb0 = jlo >> 4, kb1 = jhi >> 4;
  const int nkb = kb1 - kb0 + 1;            // <= 15 key blocks
  const int ncol = nkb * 16;                // <= 240
  const int ncol_pad = ((nkb + 1) & ~1) * 16;

  const unsigned short* Qp = Q  + (size_t)bh * SEQ * HD;
  const unsigned short* Kp = Km + (size_t)bh * SEQ * HD;
  const unsigned short* Vp = Vt + (size_t)bh * HD * SEQ;

  // Hoisted Q fragments: Hd=64 -> two K32 steps
  v16bf qa[2];
  {
    const unsigned short* qrow = Qp + (size_t)(q0 + row) * HD;
#pragma unroll
    for (int t = 0; t < 2; ++t) {
      v8bf lo = *(const v8bf*)(qrow + t * 32 + half * 16);
      v8bf hi = *(const v8bf*)(qrow + t * 32 + half * 16 + 8);
#pragma unroll
      for (int i = 0; i < 8; ++i) { qa[t][i] = lo[i]; qa[t][8 + i] = hi[i]; }
    }
  }

  // ---- scores over the window ----
  for (int kb = 0; kb < nkb; ++kb) {
    const int key0 = (kb0 + kb) * 16;
    v8f acc = {};
#pragma unroll
    for (int t = 0; t < 2; ++t) {
      v16bf kf = *(const v16bf*)(Kp + (size_t)(key0 + row) * HD + t * 32 + half * 16);
      acc = __builtin_amdgcn_wmma_f32_16x16x32_bf16(
          false, qa[t], false, kf, (short)0, acc, false, false);
    }
    const int j = key0 + row;               // key column held by this lane
#pragma unroll
    for (int r = 0; r < 8; ++r) {
      const int qi = q0 + r + half * 8;
      float v = acc[r] * 0.125f;            // 1/sqrt(64)
      if (j < qi - WHALF || j > qi + WHALF) v = -3.0e38f;
      sc[(r + half * 8) * 256 + kb * 16 + row] = v;
    }
  }

  // ---- softmax: two lanes per row (lane, lane+16 split columns) ----
  {
    const int hlen = ncol >> 1;
    const int c0   = half * hlen;
    float* srow = sc + row * 256;
    float mx = -3.0e38f;
    for (int c = 0; c < hlen; ++c) mx = fmaxf(mx, srow[c0 + c]);
    mx = fmaxf(mx, __shfl_xor(mx, 16, 32));
    float sum = 0.f;
    for (int c = 0; c < hlen; ++c) sum += __expf(srow[c0 + c] - mx);
    sum += __shfl_xor(sum, 16, 32);
    const float inv = 1.0f / sum;
    unsigned short* prow = pr + row * 256;
    for (int c = 0; c < hlen; ++c)
      prow[c0 + c] = f2bf(__expf(srow[c0 + c] - mx) * inv);
    for (int c = ncol + half; c < ncol_pad; c += 2) prow[c] = 0;  // zero pad block
  }

  // ---- attn * V : 4 N-tiles of 16 over Hd ----
  v8f oacc[4] = {v8f{}, v8f{}, v8f{}, v8f{}};
  for (int ks = 0; ks < ncol_pad; ks += 32) {
    const unsigned short* par = pr + row * 256 + ks;
    v8bf lo = *(const v8bf*)(par + half * 16);
    v8bf hi = *(const v8bf*)(par + half * 16 + 8);
    v16bf pa;
#pragma unroll
    for (int i = 0; i < 8; ++i) { pa[i] = lo[i]; pa[8 + i] = hi[i]; }

    int keyb = kb0 * 16 + ks + half * 16;
    if (keyb > SEQ - 16) keyb = SEQ - 16;   // pad block clamp (A is zero there)
#pragma unroll
    for (int i = 0; i < 4; ++i) {
      v16bf vb = *(const v16bf*)(Vp + (size_t)(i * 16 + row) * SEQ + keyb);
      oacc[i] = __builtin_amdgcn_wmma_f32_16x16x32_bf16(
          false, pa, false, vb, (short)0, oacc[i], false, false);
    }
  }

#pragma unroll
  for (int i = 0; i < 4; ++i) {
    const int dim = i * 16 + row;
#pragma unroll
    for (int r = 0; r < 8; ++r) {
      const int q = q0 + r + half * 8;
      O[((size_t)b * SEQ + q) * DM + h * HD + dim] = f2bf(oacc[i][r]);
    }
  }
}

// ---------------- host launch ----------------
extern "C" void kernel_launch(void* const* d_in, const int* in_sizes, int n_in,
                              void* d_out, int out_size, void* d_ws, size_t ws_size,
                              hipStream_t stream) {
  (void)in_sizes; (void)n_in; (void)out_size; (void)ws_size;
  const float* x  = (const float*)d_in[0];
  const float* Wq = (const float*)d_in[1];
  const float* bq = (const float*)d_in[2];
  const float* Wk = (const float*)d_in[3];
  const float* bk = (const float*)d_in[4];
  const float* Wv = (const float*)d_in[5];
  const float* bv = (const float*)d_in[6];
  const float* Wo = (const float*)d_in[7];
  const float* bo = (const float*)d_in[8];
  // d_in[9] = sparse_mask: local window is analytic, never read.

  char* ws = (char*)d_ws;
  const size_t XB = (size_t)MTOT * DM * 2;              // 8 MB
  const size_t WB = (size_t)DM * DM * 2;                // 2 MB each
  const size_t QB = (size_t)BATCH * NH * SEQ * HD * 2;  // 8 MB each
  unsigned short* xbf = (unsigned short*)(ws);
  unsigned short* wqb = (unsigned short*)(ws + XB);
  unsigned short* wkb = (unsigned short*)(ws + XB + WB);
  unsigned short* wvb = (unsigned short*)(ws + XB + 2 * WB);
  unsigned short* wob = (unsigned short*)(ws + XB + 3 * WB);
  unsigned short* Qbf = (unsigned short*)(ws + XB + 4 * WB);
  unsigned short* Kbf = (unsigned short*)(ws + XB + 4 * WB + QB);
  unsigned short* Vtb = (unsigned short*)(ws + XB + 4 * WB + 2 * QB);
  unsigned short* Obf = (unsigned short*)(ws + XB + 4 * WB + 3 * QB);

  {
    const int nx = MTOT * DM;
    cvt_f32_bf16<<<(nx / 4 + 255) / 256, 256, 0, stream>>>(x, xbf, nx);
    const int nw = DM * DM;
    const int gb = (nw / 4 + 255) / 256;
    cvt_f32_bf16<<<gb, 256, 0, stream>>>(Wq, wqb, nw);
    cvt_f32_bf16<<<gb, 256, 0, stream>>>(Wk, wkb, nw);
    cvt_f32_bf16<<<gb, 256, 0, stream>>>(Wv, wvb, nw);
    cvt_f32_bf16<<<gb, 256, 0, stream>>>(Wo, wob, nw);
  }

  // QKV projections (32 m-superblocks x 16 n-tiles = 512 workgroups of 8 waves)
  gemm16_bf16<<<512, 256, 0, stream>>>(xbf, wqb, bq, (void*)Qbf, 0);
  gemm16_bf16<<<512, 256, 0, stream>>>(xbf, wkb, bk, (void*)Kbf, 1);
  gemm16_bf16<<<512, 256, 0, stream>>>(xbf, wvb, bv, (void*)Vtb, 2);

  // Local-window attention: 4096 query blocks, 2 waves per block, 48 KB dynamic LDS
  attn_local<<<2048, 64, 49152, stream>>>(Qbf, Kbf, Vtb, Obf);

  // Output projection (fp32 result)
  gemm16_bf16<<<512, 256, 0, stream>>>(Obf, wob, bo, d_out, 3);
}
